// PaiNNLayer_39307540693931
// MI455X (gfx1250) — matrix operand order repaired and checked by reading, not compiled, not run on hardware
//
#include <hip/hip_runtime.h>

// ---------------------------------------------------------------------------
// PaiNN layer for MI455X (gfx1250): bf16 WMMA matmuls, NT-streamed edge data,
// L2-resident f32 atomics for segment sums.
// ---------------------------------------------------------------------------

typedef __attribute__((ext_vector_type(16))) __bf16 v16bf;
typedef __attribute__((ext_vector_type(8)))  float  v8f;
typedef __attribute__((ext_vector_type(4)))  float  v4f;

#define HDIM 128

__device__ __forceinline__ float clip100(float x) {
    return fminf(fmaxf(x, -100.0f), 100.0f);
}

__device__ __forceinline__ v8f wmma_bf16(v16bf a, v16bf b, v8f c) {
    // D = A(16x32) * B(32x16) + C, f32 accumulate
    return __builtin_amdgcn_wmma_f32_16x16x32_bf16(
        false, a, false, b, (short)0, c, false, false);
}

// A fragment (16x32 bf16) from LDS tile, row-major, stride lda (bf16 elems).
// Layout: lane L holds row M=L%16; element i:
//   K = i + (L>=16 ? 16 : 0)
__device__ __forceinline__ v16bf load_a_frag(const __bf16* base, int lda, int lane) {
    int row = lane & 15;
    int khi = (lane >> 4) ? 16 : 0;
    const __bf16* r = base + row * lda + khi;
    v16bf a;
#pragma unroll
    for (int p = 0; p < 8; ++p) {
        int kb = 2 * p;
        a[2 * p]     = r[kb];
        a[2 * p + 1] = r[kb + 1];
    }
    return a;
}

// B fragment (32x16 bf16) straight from global transposed weights Wt[N][K]:
// lane L holds column N = col0 + L%16, K = k0 + (L>=16?16:0) .. +15 contiguous.
__device__ __forceinline__ v16bf load_b_frag(const __bf16* wt, int strideK,
                                             int col0, int k0, int lane) {
    const __bf16* p = wt + (size_t)(col0 + (lane & 15)) * strideK
                         + k0 + ((lane >> 4) ? 16 : 0);
    return *(const v16bf*)p;   // 32B contiguous, aligned
}

__device__ __forceinline__ void atomic_add_f32(float* p, float v) {
    (void)__hip_atomic_fetch_add(p, v, __ATOMIC_RELAXED, __HIP_MEMORY_SCOPE_AGENT);
}

__device__ __forceinline__ v4f ntload4(const float* p) {
    return __builtin_nontemporal_load((const v4f*)p);
}

// Fast silu: v_exp_f32 + v_rcp_f32 (avoids IEEE div_scale/NR-refine chains).
__device__ __forceinline__ float silu_f(float h) {
    return h * __builtin_amdgcn_rcpf(1.0f + __expf(-h));
}

// ---------------------------------------------------------------------------
// K0: convert weight matrix [K][Ncol] f32 -> transposed bf16 [Ncol][K]
// ---------------------------------------------------------------------------
__global__ void k_w_to_bf16t(const float* __restrict__ src, __bf16* __restrict__ dst,
                             int K, int Ncol) {
    int i = blockIdx.x * blockDim.x + threadIdx.x;
    if (i >= K * Ncol) return;
    int k = i / Ncol, n = i - k * Ncol;
    dst[(size_t)n * K + k] = (__bf16)src[i];
}

// ---------------------------------------------------------------------------
// K1: x = silu(s @ W1 + b1) @ W2 + b2   -> ws x[N][384] f32
// Block = 16 nodes, 128 threads (4 waves).
// ---------------------------------------------------------------------------
__global__ __launch_bounds__(128) void k_message_mlp(
    const float* __restrict__ s, const float* __restrict__ b1,
    const float* __restrict__ b2, const __bf16* __restrict__ W1t,
    const __bf16* __restrict__ W2t, float* __restrict__ x, int n_nodes) {
    __shared__ __bf16 s_bf[16][132];
    __shared__ __bf16 hid_bf[16][132];
    int node0 = blockIdx.x * 16;
    int tid = threadIdx.x, lane = tid & 31, wave = tid >> 5;

    for (int i = tid; i < 16 * HDIM; i += 128) {
        int r = i >> 7, c = i & 127;
        int g = node0 + r;
        float val = (g < n_nodes) ? s[(size_t)g * HDIM + c] : 0.0f;
        s_bf[r][c] = (__bf16)val;
    }
    __syncthreads();

    // hidden = silu(s @ W1 + b1): 8 column tiles
    for (int t = wave; t < 8; t += 4) {
        int col0 = t * 16;
        v8f acc = {};
#pragma unroll
        for (int ks = 0; ks < 4; ++ks) {
            v16bf a = load_a_frag(&s_bf[0][0] + ks * 32, 132, lane);
            v16bf b = load_b_frag(W1t, 128, col0, ks * 32, lane);
            acc = wmma_bf16(a, b, acc);
        }
        int col = col0 + (lane & 15);
        float bias = b1[col];
        int rbase = (lane >> 4) ? 8 : 0;
#pragma unroll
        for (int i = 0; i < 8; ++i)
            hid_bf[rbase + i][col] = (__bf16)silu_f(acc[i] + bias);
    }
    __syncthreads();

    // x = hidden @ W2 + b2: 24 column tiles
    for (int t = wave; t < 24; t += 4) {
        int col0 = t * 16;
        v8f acc = {};
#pragma unroll
        for (int ks = 0; ks < 4; ++ks) {
            v16bf a = load_a_frag(&hid_bf[0][0] + ks * 32, 132, lane);
            v16bf b = load_b_frag(W2t, 128, col0, ks * 32, lane);
            acc = wmma_bf16(a, b, acc);
        }
        int col = col0 + (lane & 15);
        float bias = b2[col];
        int rbase = (lane >> 4) ? 8 : 0;
#pragma unroll
        for (int i = 0; i < 8; ++i) {
            int g = node0 + rbase + i;
            if (g < n_nodes) x[(size_t)g * 384 + col] = acc[i] + bias;
        }
    }
}

// ---------------------------------------------------------------------------
// K2: edge scatter. One wave per edge. Wij streamed non-temporally (1.23 GB);
// x / v gathers stay cached (L2 resident). f32 atomics into accumulators.
// ---------------------------------------------------------------------------
__global__ __launch_bounds__(256) void k_edges(
    const float* __restrict__ x, const float* __restrict__ Wij,
    const float* __restrict__ dir_ij, const float* __restrict__ v,
    const int* __restrict__ senders, const int* __restrict__ receivers,
    float* __restrict__ ds_acc, float* __restrict__ dv_acc, int n_edges) {
    int e = (int)((blockIdx.x * (unsigned)blockDim.x + threadIdx.x) >> 5);
    if (e >= n_edges) return;
    int lane = threadIdx.x & 31;

    int recv = receivers[e];
    int send = senders[e];
    const float* we = Wij + (size_t)e * 384;
    const float* xr = x + (size_t)recv * 384;
    const float* vr = v + (size_t)recv * 384;
    int c = lane * 4;

    v4f w0 = ntload4(we + c);
    v4f w1 = ntload4(we + 128 + c);
    v4f w2 = ntload4(we + 256 + c);
    v4f x0 = *(const v4f*)(xr + c);
    v4f x1 = *(const v4f*)(xr + 128 + c);
    v4f x2 = *(const v4f*)(xr + 256 + c);

    v4f ds  = w0 * x0;
    v4f dv1 = w1 * x1;
    v4f dv2 = w2 * x2;

    float* dsp = ds_acc + (size_t)send * HDIM + c;
#pragma unroll
    for (int i = 0; i < 4; ++i) atomic_add_f32(dsp + i, ds[i]);

    const float* de = dir_ij + (size_t)e * 3;
#pragma unroll
    for (int a = 0; a < 3; ++a) {
        float d = de[a];
        v4f vv = *(const v4f*)(vr + a * HDIM + c);
        v4f dv = dv1 * d + dv2 * vv;
        float* dvp = dv_acc + (size_t)send * 384 + a * HDIM + c;
#pragma unroll
        for (int i = 0; i < 4; ++i) atomic_add_f32(dvp + i, dv[i]);
    }
}

// ---------------------------------------------------------------------------
// K3: intra-particle update. Block = 16 nodes, 128 threads (4 waves).
// LDS unioned to 62,720 B (< 64 KB static limit):
//   [0)      vl_bf  [48][132]             12672
//   [12672)  vr_bf  [48][132]             12672
//   [25344)  spost  [16][130] f32          8320
//   [33664)  union region (29056):
//            stage1: vp_bf [48][132]
//            stage3: hid @33664 [16][132], ts @37888 [16][264]
//            stage4: hid @33664,          u  @37888 [16][388] f32
// ---------------------------------------------------------------------------
__global__ __launch_bounds__(128) void k_update(
    const float* __restrict__ s, const float* __restrict__ v,
    const float* __restrict__ ds_acc, const float* __restrict__ dv_acc,
    const __bf16* __restrict__ Wvt, const __bf16* __restrict__ Wm1t,
    const __bf16* __restrict__ Wm2t, const float* __restrict__ bm1,
    const float* __restrict__ bm2, float* __restrict__ out_s,
    float* __restrict__ out_v, int n_nodes) {
    __shared__ __align__(16) unsigned char smem[62720];
    __bf16* vl    = (__bf16*)smem;                    // [48][132]
    __bf16* vrr   = (__bf16*)(smem + 12672);          // [48][132]
    float*  spost = (float*)(smem + 25344);           // [16][130]
    __bf16* vp    = (__bf16*)(smem + 33664);          // [48][132]  (stage 1)
    __bf16* hid   = (__bf16*)(smem + 33664);          // [16][132]  (stages 3-4)
    __bf16* ts    = (__bf16*)(smem + 37888);          // [16][264]  (stages 2-3)
    float*  u     = (float*)(smem + 37888);           // [16][388]  (stages 4-5)

    int node0 = blockIdx.x * 16;
    int tid = threadIdx.x, lane = tid & 31, wave = tid >> 5;

    // ---- stage 0: s_post = s + clip(ds);  vp = v + clip(dv) (bf16) ----
    for (int i = tid; i < 16 * HDIM; i += 128) {
        int r = i >> 7, c = i & 127;
        int g = node0 + r;
        float val = 0.0f;
        if (g < n_nodes) {
            size_t gi = (size_t)g * HDIM + c;
            val = s[gi] + clip100(ds_acc[gi]);
        }
        spost[r * 130 + c] = val;
    }
    for (int i = tid; i < 48 * HDIM; i += 128) {
        int r = i >> 7, c = i & 127;          // r = node*3 + axis
        float val = 0.0f;
        if (node0 + r / 3 < n_nodes) {
            size_t gi = (size_t)(node0 * 3 + r) * HDIM + c;
            val = v[gi] + clip100(dv_acc[gi]);
        }
        vp[r * 132 + c] = (__bf16)val;
    }
    __syncthreads();

    // ---- stage 1: [48,256] = vp @ Wv ; split into vl / vr ----
    for (int t = wave; t < 48; t += 4) {
        int rt = t >> 4, ct = t & 15;
        v8f acc = {};
#pragma unroll
        for (int ks = 0; ks < 4; ++ks) {
            v16bf a = load_a_frag(vp + rt * 16 * 132 + ks * 32, 132, lane);
            v16bf b = load_b_frag(Wvt, 128, ct * 16, ks * 32, lane);
            acc = wmma_bf16(a, b, acc);
        }
        int col = ct * 16 + (lane & 15);
        __bf16* dst = (col < HDIM) ? vl : vrr;
        int cc = col & 127;
        int rbase = rt * 16 + ((lane >> 4) ? 8 : 0);
#pragma unroll
        for (int i = 0; i < 8; ++i)
            dst[(rbase + i) * 132 + cc] = (__bf16)acc[i];
    }
    __syncthreads();

    // ---- stage 2: ts = [s_post | ||v_r||] (bf16) ----
    for (int i = tid; i < 16 * HDIM; i += 128) {
        int n = i >> 7, c = i & 127;
        float a0 = (float)vrr[(n * 3 + 0) * 132 + c];
        float a1 = (float)vrr[(n * 3 + 1) * 132 + c];
        float a2 = (float)vrr[(n * 3 + 2) * 132 + c];
        float nrm = __builtin_amdgcn_sqrtf(a0 * a0 + a1 * a1 + a2 * a2 + 1e-8f);
        ts[n * 264 + c]        = (__bf16)spost[n * 130 + c];
        ts[n * 264 + HDIM + c] = (__bf16)nrm;
    }
    __syncthreads();

    // ---- stage 3: hid = silu(ts @ Wm1 + bm1) ----
    for (int t = wave; t < 8; t += 4) {
        int col0 = t * 16;
        v8f acc = {};
#pragma unroll
        for (int ks = 0; ks < 8; ++ks) {
            v16bf a = load_a_frag(ts + ks * 32, 264, lane);
            v16bf b = load_b_frag(Wm1t, 256, col0, ks * 32, lane);
            acc = wmma_bf16(a, b, acc);
        }
        int col = col0 + (lane & 15);
        float bias = bm1[col];
        int rbase = (lane >> 4) ? 8 : 0;
#pragma unroll
        for (int i = 0; i < 8; ++i)
            hid[(rbase + i) * 132 + col] = (__bf16)silu_f(acc[i] + bias);
    }
    __syncthreads();

    // ---- stage 4: u = hid @ Wm2 + bm2 (f32 in LDS) ----
    for (int t = wave; t < 24; t += 4) {
        int col0 = t * 16;
        v8f acc = {};
#pragma unroll
        for (int ks = 0; ks < 4; ++ks) {
            v16bf a = load_a_frag(hid + ks * 32, 132, lane);
            v16bf b = load_b_frag(Wm2t, 128, col0, ks * 32, lane);
            acc = wmma_bf16(a, b, acc);
        }
        int col = col0 + (lane & 15);
        float bias = bm2[col];
        int rbase = (lane >> 4) ? 8 : 0;
#pragma unroll
        for (int i = 0; i < 8; ++i)
            u[(rbase + i) * 388 + col] = acc[i] + bias;
    }
    __syncthreads();

    // ---- stage 5: final elementwise combine + store outputs ----
    for (int i = tid; i < 16 * HDIM; i += 128) {
        int n = i >> 7, c = i & 127;
        int g = node0 + n;
        if (g >= n_nodes) continue;
        float dot = 0.0f;
#pragma unroll
        for (int a = 0; a < 3; ++a)
            dot += (float)vl[(n * 3 + a) * 132 + c] * (float)vrr[(n * 3 + a) * 132 + c];
        float val = u[n * 388 + c] + u[n * 388 + 256 + c] * dot;
        out_s[(size_t)g * HDIM + c] = spost[n * 130 + c] + clip100(val);
    }
    for (int i = tid; i < 48 * HDIM; i += 128) {
        int r = i >> 7, c = i & 127;
        int n = r / 3;
        if (node0 + n >= n_nodes) continue;
        size_t gi = (size_t)(node0 * 3 + r) * HDIM + c;
        float vpv  = v[gi] + clip100(dv_acc[gi]);
        float term = (float)vl[r * 132 + c] * u[n * 388 + HDIM + c];
        out_v[gi] = vpv + clip100(term);
    }
}

// ---------------------------------------------------------------------------
extern "C" void kernel_launch(void* const* d_in, const int* in_sizes, int n_in,
                              void* d_out, int out_size, void* d_ws, size_t ws_size,
                              hipStream_t stream) {
    const float* s      = (const float*)d_in[0];
    const float* v      = (const float*)d_in[1];
    const float* dir_ij = (const float*)d_in[2];
    const float* Wij    = (const float*)d_in[3];
    const int*   send   = (const int*)d_in[4];
    const int*   recv   = (const int*)d_in[5];
    const float* W1 = (const float*)d_in[6],  *b1  = (const float*)d_in[7];
    const float* W2 = (const float*)d_in[8],  *b2  = (const float*)d_in[9];
    const float* Wm1 = (const float*)d_in[10], *bm1 = (const float*)d_in[11];
    const float* Wm2 = (const float*)d_in[12], *bm2 = (const float*)d_in[13];
    const float* Wv = (const float*)d_in[14];

    const int H = HDIM;
    const int N = in_sizes[0] / H;
    const int E = in_sizes[2] / 3;

    unsigned char* ws = (unsigned char*)d_ws;
    size_t off = 0;
    auto alloc = [&](size_t bytes) {
        off = (off + 255) & ~(size_t)255;
        size_t o = off; off += bytes; return o;
    };
    float*  x_ws  = (float*)(ws + alloc((size_t)N * 384 * 4));
    float*  ds_ws = (float*)(ws + alloc((size_t)N * 128 * 4));
    float*  dv_ws = (float*)(ws + alloc((size_t)N * 384 * 4));
    __bf16* W1t  = (__bf16*)(ws + alloc(128 * 128 * 2));
    __bf16* W2t  = (__bf16*)(ws + alloc(128 * 384 * 2));
    __bf16* Wm1t = (__bf16*)(ws + alloc(256 * 128 * 2));
    __bf16* Wm2t = (__bf16*)(ws + alloc(128 * 384 * 2));
    __bf16* Wvt  = (__bf16*)(ws + alloc(128 * 256 * 2));

    float* out_s = (float*)d_out;
    float* out_v = out_s + (size_t)N * 128;

    // zero segment-sum accumulators (graph-capturable)
    hipMemsetAsync(ds_ws, 0, (size_t)N * 128 * 4, stream);
    hipMemsetAsync(dv_ws, 0, (size_t)N * 384 * 4, stream);

    // weights -> transposed bf16
    auto tgrid = [](int n) { return (n + 255) / 256; };
    k_w_to_bf16t<<<tgrid(128 * 128), 256, 0, stream>>>(W1, W1t, 128, 128);
    k_w_to_bf16t<<<tgrid(128 * 384), 256, 0, stream>>>(W2, W2t, 128, 384);
    k_w_to_bf16t<<<tgrid(256 * 128), 256, 0, stream>>>(Wm1, Wm1t, 256, 128);
    k_w_to_bf16t<<<tgrid(128 * 384), 256, 0, stream>>>(Wm2, Wm2t, 128, 384);
    k_w_to_bf16t<<<tgrid(128 * 256), 256, 0, stream>>>(Wv, Wvt, 128, 256);

    int nblk = (N + 15) / 16;
    k_message_mlp<<<nblk, 128, 0, stream>>>(s, b1, b2, W1t, W2t, x_ws, N);

    long long threads = (long long)E * 32;
    int eblk = (int)((threads + 255) / 256);
    k_edges<<<eblk, 256, 0, stream>>>(x_ws, Wij, dir_ij, v, send, recv,
                                      ds_ws, dv_ws, E);

    k_update<<<nblk, 128, 0, stream>>>(s, v, ds_ws, dv_ws, Wvt, Wm1t, Wm2t,
                                       bm1, bm2, out_s, out_v, N);
}
